// TetrahedralAttention_12489764897459
// MI455X (gfx1250) — matrix operand order repaired and checked by reading, not compiled, not run on hardware
//
#include <hip/hip_runtime.h>

// ---------------------------------------------------------------------------
// TetrahedralAttention for MI455X (gfx1250, wave32, WMMA 16x16x32 f16->f32)
//   HIDDEN=256, HEADS=8, HEAD_DIM=32, B=4, S=2048
// Pipeline:
//   k0: one-shot f32->f16 conversion of x and Wq/Wk/Wv/Wo into workspace
//       (removes redundant per-tile v_cvt work and halves GEMM read bytes).
//   k1: Q/K/V projections, f16 GEMM, wave computes 16x64 strip (A reuse x4).
//       Q,K stored [b,h,s,d]; V stored transposed [b,h,d,s].
//   k2: flash attention per 16-query tile: scores WMMA, online softmax,
//       P->LDS->A-frag, P@V WMMA, global_prefetch of next K/V blocks.
//   k3: output projection (attn @ Wo^T), 16x64 strips -> f32 d_out.
// ---------------------------------------------------------------------------

typedef __attribute__((ext_vector_type(16))) _Float16 v16h;
typedef __attribute__((ext_vector_type(8)))  float    v8f;

union Frag16 {
    v16h     v;
    _Float16 h[16];
    uint4    q[2];
};

#define HIDDEN   256
#define HEADS    8
#define HEAD_DIM 32
#define BATCH    4
#define SEQ      2048
#define ROWS     (BATCH * SEQ)          // 8192
#define NX       (ROWS * HIDDEN)        // 2,097,152 x elements
#define NW       (HIDDEN * HIDDEN)      // 65,536 per weight

static __device__ __forceinline__ v8f wmma_f16(const Frag16& a, const Frag16& b, v8f c) {
    return __builtin_amdgcn_wmma_f32_16x16x32_f16(
        /*neg_a=*/false, a.v, /*neg_b=*/false, b.v,
        /*c_mod=*/(short)0, c, /*reuse_a=*/false, /*reuse_b=*/false);
}

// ---------------------------------------------------------------------------
// Kernel 0: convert x and the four weight matrices to f16 once.
//   (NX + 4*NW) / 8 per-thread chunks = 294912 threads -> 1152 blocks.
// ---------------------------------------------------------------------------
__global__ void __launch_bounds__(256)
ta_convert(const float* __restrict__ x,
           const float* __restrict__ Wq, const float* __restrict__ Wk,
           const float* __restrict__ Wv, const float* __restrict__ Wo,
           _Float16* __restrict__ xh,    // [ROWS, HIDDEN]
           _Float16* __restrict__ wh)    // [Wq|Wk|Wv|Wo] each [256,256]
{
    const size_t off = ((size_t)blockIdx.x * 256 + threadIdx.x) * 8;
    const float* src;
    _Float16* dst;
    if (off < NX) {
        src = x + off;
        dst = xh + off;
    } else {
        const size_t r = off - NX;
        const int w = (int)(r >> 16);          // which weight
        const size_t wi = r & (NW - 1);
        src = ((w == 0) ? Wq : (w == 1) ? Wk : (w == 2) ? Wv : Wo) + wi;
        dst = wh + r;
    }
    const float4 a = *(const float4*)(src);
    const float4 b = *(const float4*)(src + 4);
    Frag16 o;   // use low 8 halves
    o.h[0] = (_Float16)a.x; o.h[1] = (_Float16)a.y;
    o.h[2] = (_Float16)a.z; o.h[3] = (_Float16)a.w;
    o.h[4] = (_Float16)b.x; o.h[5] = (_Float16)b.y;
    o.h[6] = (_Float16)b.z; o.h[7] = (_Float16)b.w;
    *(uint4*)dst = o.q[0];
}

// ---------------------------------------------------------------------------
// Kernel 1: QKV projection, f16 in / f16 out. One wave per 16x64 strip.
//   waves = 3 * 512 row-tiles * 4 col-groups = 6144 -> 768 blocks.
// ---------------------------------------------------------------------------
__global__ void __launch_bounds__(256)
ta_qkv_proj(const _Float16* __restrict__ xh,
            const _Float16* __restrict__ wh,
            _Float16* __restrict__ qout,    // [B,H,S,D]
            _Float16* __restrict__ kout,    // [B,H,S,D]
            _Float16* __restrict__ vtout)   // [B,H,D,S]
{
    const int lane = threadIdx.x & 31;
    const int widx = threadIdx.x >> 5;
    const int wg   = blockIdx.x * 8 + widx;

    const int which = wg / 2048;            // 0=Q 1=K 2=V
    const int rem   = wg - which * 2048;
    const int rt    = rem >> 2;             // row tile [0,512)
    const int cg    = rem & 3;              // 64-col group [0,4)

    const _Float16* W = wh + (size_t)which * NW;

    const int m  = lane & 15;
    const int hi = lane >> 4;

    v8f acc[4] = {{}, {}, {}, {}};
    #pragma unroll
    for (int k0 = 0; k0 < HIDDEN; k0 += 32) {
        Frag16 a;
        const _Float16* ap = xh + (size_t)(rt * 16 + m) * HIDDEN + k0;
        a.q[0] = *(const uint4*)(ap + hi * 16);
        a.q[1] = *(const uint4*)(ap + hi * 16 + 8);
        #pragma unroll
        for (int c = 0; c < 4; ++c) {
            Frag16 b;
            const _Float16* bp = W + (size_t)(cg * 64 + c * 16 + m) * HIDDEN + k0 + hi * 16;
            b.q[0] = *(const uint4*)(bp);
            b.q[1] = *(const uint4*)(bp + 8);
            acc[c] = wmma_f16(a, b, acc[c]);
        }
    }

    #pragma unroll
    for (int c = 0; c < 4; ++c) {
        const int col = cg * 64 + c * 16 + (lane & 15);
        const int h   = col >> 5;
        const int d   = col & 31;
        #pragma unroll
        for (int j = 0; j < 8; ++j) {
            const int row = rt * 16 + j + hi * 8;    // b*S + s
            const int b   = row >> 11;
            const int s   = row & (SEQ - 1);
            const _Float16 val = (_Float16)acc[c][j];
            if (which == 2) {
                vtout[((size_t)(b * HEADS + h) * HEAD_DIM + d) * SEQ + s] = val;
            } else {
                _Float16* o = (which == 0) ? qout : kout;
                o[((size_t)(b * HEADS + h) * SEQ + s) * HEAD_DIM + d] = val;
            }
        }
    }
}

// ---------------------------------------------------------------------------
// Kernel 2: flash attention. One wave per (b,h,16-query tile).
//   waves = 32 bh * 128 qtiles = 4096  -> 512 blocks of 8 waves.
// ---------------------------------------------------------------------------
__global__ void __launch_bounds__(256)
ta_flash_attn(const _Float16* __restrict__ Qb,   // [B,H,S,D]
              const _Float16* __restrict__ Kb,   // [B,H,S,D]
              const _Float16* __restrict__ Vtb,  // [B,H,D,S]
              const float* __restrict__ geo,     // [S,S]
              const int*   __restrict__ amask,   // [B,S]
              const float* __restrict__ head_gates,
              const float* __restrict__ geo_scale_p,
              const float* __restrict__ head_geo_scales,
              _Float16* __restrict__ attn_out)   // [B,S,HIDDEN] f16
{
    __shared__ __align__(16) _Float16 pbuf[8 * 16 * 32];  // per-wave P tile

    const int lane = threadIdx.x & 31;
    const int widx = threadIdx.x >> 5;
    const int wg   = blockIdx.x * 8 + widx;

    const int bh = wg >> 7;              // [0,32)
    const int qt = wg & 127;             // [0,128)
    const int b  = bh >> 3;
    const int h  = bh & 7;
    const int qbase = qt * 16;

    const int n  = lane & 15;            // column / M-row index within tile
    const int hi = lane >> 4;

    const _Float16* Qh = Qb  + (size_t)bh * SEQ * HEAD_DIM;
    const _Float16* Kh = Kb  + (size_t)bh * SEQ * HEAD_DIM;
    const _Float16* Vh = Vtb + (size_t)bh * HEAD_DIM * SEQ;

    // Q A-fragment: lane holds query row qbase+n, K(=d) runs per half-wave
    Frag16 qf;
    {
        const _Float16* qp = Qh + (size_t)(qbase + n) * HEAD_DIM;
        const int base = hi * 16;
        qf.q[0] = *(const uint4*)(qp + base);
        qf.q[1] = *(const uint4*)(qp + base + 8);
    }

    const float invs = 0.17677669529663687f;   // 1/sqrt(32)
    const float gs   = geo_scale_p[0] * head_geo_scales[h];

    float m_i[8], l_i[8];
    #pragma unroll
    for (int j = 0; j < 8; ++j) { m_i[j] = -1e30f; l_i[j] = 0.0f; }
    v8f acc0 = {}, acc1 = {};

    _Float16* pw = pbuf + widx * (16 * 32);

    for (int kb = 0; kb < SEQ; kb += 32) {
        // K B-fragments: lane = key column, d = e + 16*hi (contiguous)
        const _Float16* kp0 = Kh + (size_t)(kb + n) * HEAD_DIM + hi * 16;
        const _Float16* kp1 = Kh + (size_t)(kb + 16 + n) * HEAD_DIM + hi * 16;
        Frag16 kf0, kf1;
        kf0.q[0] = *(const uint4*)(kp0);
        kf0.q[1] = *(const uint4*)(kp0 + 8);
        kf1.q[0] = *(const uint4*)(kp1);
        kf1.q[1] = *(const uint4*)(kp1 + 8);

        // prefetch next key block (K rows + V^T rows) -> global_prefetch_b8
        if (kb + 32 < SEQ) {
            __builtin_prefetch(kp0 + 32 * HEAD_DIM, 0, 3);
            __builtin_prefetch(kp1 + 32 * HEAD_DIM, 0, 3);
            __builtin_prefetch(Vh + (size_t)n * SEQ + kb + 32 + hi * 16, 0, 3);
            __builtin_prefetch(Vh + (size_t)(n + 16) * SEQ + kb + 32 + hi * 16, 0, 3);
        }

        v8f z = {};
        v8f s0 = wmma_f16(qf, kf0, z);
        v8f s1 = wmma_f16(qf, kf1, z);

        const int key0 = kb + n;
        const int key1 = kb + 16 + n;
        const int ms0 = amask[b * SEQ + key0];
        const int ms1 = amask[b * SEQ + key1];

        #pragma unroll
        for (int j = 0; j < 8; ++j) {
            const int q = qbase + j + hi * 8;
            const float b0 = gs * geo[(size_t)q * SEQ + key0];
            const float b1 = gs * geo[(size_t)q * SEQ + key1];
            float v0 = (ms0 != 0) ? (s0[j] * invs + b0) : -1e30f;
            float v1 = (ms1 != 0) ? (s1[j] * invs + b1) : -1e30f;

            // row max over 16 lanes of this half-wave
            float mx = fmaxf(v0, v1);
            #pragma unroll
            for (int off = 1; off < 16; off <<= 1)
                mx = fmaxf(mx, __shfl_xor(mx, off, 32));
            const float mnew = fmaxf(m_i[j], mx);
            const float resc = __expf(m_i[j] - mnew);
            m_i[j] = mnew;

            const float p0 = __expf(v0 - mnew);
            const float p1 = __expf(v1 - mnew);
            float rs = p0 + p1;
            #pragma unroll
            for (int off = 1; off < 16; off <<= 1)
                rs += __shfl_xor(rs, off, 32);
            l_i[j] = l_i[j] * resc + rs;

            acc0[j] *= resc;
            acc1[j] *= resc;

            // stash P tile row-major [16 q][32 k] as f16 (same-wave LDS)
            const int prow = j + hi * 8;
            pw[prow * 32 + n]      = (_Float16)p0;
            pw[prow * 32 + 16 + n] = (_Float16)p1;
        }

        // reload P as A-fragment (lane = q row, k runs per half-wave)
        Frag16 pf;
        {
            const _Float16* pp = pw + n * 32;
            const int base = hi * 16;
            pf.q[0] = *(const uint4*)(pp + base);
            pf.q[1] = *(const uint4*)(pp + base + 8);
        }
        // V B-fragments from transposed layout: lane = d column, key contiguous
        Frag16 vf0, vf1;
        {
            const _Float16* vp0 = Vh + (size_t)n * SEQ + kb + hi * 16;
            vf0.q[0] = *(const uint4*)(vp0);
            vf0.q[1] = *(const uint4*)(vp0 + 8);
            const _Float16* vp1 = Vh + (size_t)(n + 16) * SEQ + kb + hi * 16;
            vf1.q[0] = *(const uint4*)(vp1);
            vf1.q[1] = *(const uint4*)(vp1 + 8);
        }
        acc0 = wmma_f16(pf, vf0, acc0);
        acc1 = wmma_f16(pf, vf1, acc1);
    }

    // finalize: out = gate * acc / l ; write [b,s,h*32+d] f16
    const float gate = head_gates[h];
    #pragma unroll
    for (int j = 0; j < 8; ++j) {
        const float norm = gate / l_i[j];
        const int q = qbase + j + hi * 8;
        _Float16* orow = attn_out + ((size_t)b * SEQ + q) * HIDDEN + h * HEAD_DIM;
        orow[n]      = (_Float16)(acc0[j] * norm);
        orow[16 + n] = (_Float16)(acc1[j] * norm);
    }
}

// ---------------------------------------------------------------------------
// Kernel 3: output projection attn @ Wo^T -> f32 d_out. 16x64 strips.
//   waves = 512 row tiles * 4 col groups = 2048 -> 256 blocks.
// ---------------------------------------------------------------------------
__global__ void __launch_bounds__(256)
ta_out_proj(const _Float16* __restrict__ attn,  // [ROWS, HIDDEN] f16
            const _Float16* __restrict__ woh,   // [256,256] f16
            float* __restrict__ out)            // [ROWS, HIDDEN] f32
{
    const int lane = threadIdx.x & 31;
    const int widx = threadIdx.x >> 5;
    const int wg   = blockIdx.x * 8 + widx;

    const int rt = wg >> 2;
    const int cg = wg & 3;

    const int m  = lane & 15;
    const int hi = lane >> 4;

    v8f acc[4] = {{}, {}, {}, {}};
    #pragma unroll
    for (int k0 = 0; k0 < HIDDEN; k0 += 32) {
        Frag16 a;
        const _Float16* ap = attn + (size_t)(rt * 16 + m) * HIDDEN + k0;
        a.q[0] = *(const uint4*)(ap + hi * 16);
        a.q[1] = *(const uint4*)(ap + hi * 16 + 8);
        #pragma unroll
        for (int c = 0; c < 4; ++c) {
            Frag16 b;
            const _Float16* bp = woh + (size_t)(cg * 64 + c * 16 + m) * HIDDEN + k0 + hi * 16;
            b.q[0] = *(const uint4*)(bp);
            b.q[1] = *(const uint4*)(bp + 8);
            acc[c] = wmma_f16(a, b, acc[c]);
        }
    }

    #pragma unroll
    for (int c = 0; c < 4; ++c) {
        #pragma unroll
        for (int j = 0; j < 8; ++j) {
            const int row = rt * 16 + j + hi * 8;
            out[(size_t)row * HIDDEN + cg * 64 + c * 16 + m] = acc[c][j];
        }
    }
}

// ---------------------------------------------------------------------------
// Launch
// ---------------------------------------------------------------------------
extern "C" void kernel_launch(void* const* d_in, const int* in_sizes, int n_in,
                              void* d_out, int out_size, void* d_ws, size_t ws_size,
                              hipStream_t stream) {
    const float* x    = (const float*)d_in[0];
    const float* geo  = (const float*)d_in[1];
    const int*   mask = (const int*)d_in[2];
    const float* gate = (const float*)d_in[3];
    const float* Wq   = (const float*)d_in[4];
    const float* Wk   = (const float*)d_in[5];
    const float* Wv   = (const float*)d_in[6];
    const float* Wo   = (const float*)d_in[7];
    const float* gbs  = (const float*)d_in[8];
    const float* hgs  = (const float*)d_in[9];

    // workspace layout (f16 elements):
    //   xh   : NX            (x in f16)
    //   wh   : 4*NW          (Wq|Wk|Wv|Wo in f16)
    //   q    : NX            ([B,H,S,D])
    //   k    : NX            ([B,H,S,D])
    //   vt   : NX            ([B,H,D,S])
    //   attn : NX            ([B,S,HIDDEN])
    _Float16* xh   = (_Float16*)d_ws;
    _Float16* wh   = xh + (size_t)NX;
    _Float16* qws  = wh + (size_t)4 * NW;
    _Float16* kws  = qws + (size_t)NX;
    _Float16* vtws = kws + (size_t)NX;
    _Float16* aws  = vtws + (size_t)NX;

    // k0: (NX + 4*NW)/8 elements per thread -> 1152 blocks of 256
    ta_convert<<<1152, 256, 0, stream>>>(x, Wq, Wk, Wv, Wo, xh, wh);
    // k1: 6144 waves / 8 = 768 blocks
    ta_qkv_proj<<<768, 256, 0, stream>>>(xh, wh, qws, kws, vtws);
    // k2: 4096 waves / 8 = 512 blocks
    ta_flash_attn<<<512, 256, 0, stream>>>(qws, kws, vtws, geo, mask,
                                           gate, gbs, hgs, aws);
    // k3: 2048 waves / 8 = 256 blocks
    ta_out_proj<<<256, 256, 0, stream>>>(aws, wh + (size_t)3 * NW, (float*)d_out);
}
